// SMPNN_8701603742429
// MI455X (gfx1250) — matrix operand order repaired and checked by reading, not compiled, not run on hardware
//
#include <hip/hip_runtime.h>
#include <hip/hip_bf16.h>
#include <math.h>

#define D_ 128

typedef __attribute__((ext_vector_type(16))) __bf16 v16bf;
typedef __attribute__((ext_vector_type(8)))  float  v8f;

union BF8 { v16bf v; unsigned u[8]; };

static __device__ __forceinline__ unsigned short bf16bits(float x) {
  return __builtin_bit_cast(unsigned short, (__bf16)x);
}
static __device__ __forceinline__ unsigned packbf2(float lo, float hi) {
  return (unsigned)bf16bits(lo) | ((unsigned)bf16bits(hi) << 16);
}
static __device__ __forceinline__ float silu_f(float x) {
  return x / (1.0f + __expf(-x));
}

// ---------------- degree / norm ----------------
__global__ void deg_init_kernel(float* __restrict__ deg, int n) {
  int i = blockIdx.x * 256 + threadIdx.x;
  if (i < n) deg[i] = 1.0f;   // self-loop
}
__global__ void deg_count_kernel(const int* __restrict__ col, float* __restrict__ deg, int e) {
  int i = blockIdx.x * 256 + threadIdx.x;
  if (i < e) atomicAdd(&deg[col[i]], 1.0f);
}
__global__ void dinv_kernel(const float* __restrict__ deg, float* __restrict__ dinv, int n) {
  int i = blockIdx.x * 256 + threadIdx.x;
  if (i < n) dinv[i] = rsqrtf(deg[i]);   // deg >= 1 always (self-loop)
}

// ---------------- weight packing into WMMA B-fragment layout ----------------
// wpk[((ks*8 + nt)*32 + lane)*8 + v] = pack(bf16 W[k][n], bf16 W[k+1][n])
//   n = nt*16 + (lane&15);  k = ks*32 + ((lane>=16)?8:0) + 2*v + ((v>=4)?8:0)
__global__ void pack_w_kernel(const float* __restrict__ W, unsigned* __restrict__ out) {
  int idx = blockIdx.x * 256 + threadIdx.x;          // 0..8191
  int v    = idx & 7;
  int lane = (idx >> 3) & 31;
  int nt   = (idx >> 8) & 7;
  int ks   = (idx >> 11) & 3;
  int n = nt * 16 + (lane & 15);
  int k = ks * 32 + ((lane >> 4) ? 8 : 0) + 2 * v + ((v >= 4) ? 8 : 0);
  out[idx] = packbf2(W[k * D_ + n], W[(k + 1) * D_ + n]);
}

// ---------------- zero dwords ----------------
__global__ void zero_u32_kernel(unsigned* __restrict__ p, int total) {
  int i = blockIdx.x * 256 + threadIdx.x;
  if (i < total) p[i] = 0u;
}
__global__ void zero_f32_kernel(float* __restrict__ p, int total) {
  int i = blockIdx.x * 256 + threadIdx.x;
  if (i < total) p[i] = 0.0f;
}

// ---------------- fp32 -> bf16 elementwise ----------------
__global__ void cvt_bf16_kernel(const float* __restrict__ in, unsigned short* __restrict__ out, int total) {
  int i = blockIdx.x * 256 + threadIdx.x;
  if (i < total) out[i] = bf16bits(in[i]);
}

// ---------------- LayerNorm -> bf16 (one wave per row) ----------------
__global__ __launch_bounds__(256) void ln_bf16_kernel(
    const float* __restrict__ x, const float* __restrict__ g, const float* __restrict__ b,
    unsigned short* __restrict__ out, int nrows) {
  int wave = threadIdx.x >> 5, lane = threadIdx.x & 31;
  int row = blockIdx.x * 8 + wave;
  if (row >= nrows) return;
  const float* xr = x + (size_t)row * D_;
  float v[4];
  float s = 0.0f;
#pragma unroll
  for (int j = 0; j < 4; ++j) { v[j] = xr[lane + 32 * j]; s += v[j]; }
#pragma unroll
  for (int m = 16; m; m >>= 1) s += __shfl_xor(s, m, 32);
  float mu = s * (1.0f / 128.0f);
  float var = 0.0f;
#pragma unroll
  for (int j = 0; j < 4; ++j) { float d = v[j] - mu; var += d * d; }
#pragma unroll
  for (int m = 16; m; m >>= 1) var += __shfl_xor(var, m, 32);
  float rs = rsqrtf(var * (1.0f / 128.0f) + 1e-5f);
  unsigned short* orow = out + (size_t)row * D_;
#pragma unroll
  for (int j = 0; j < 4; ++j) {
    int c = lane + 32 * j;
    orow[c] = bf16bits((v[j] - mu) * rs * g[c] + b[c]);
  }
}

// ---------------- WMMA GEMM: [Npad x 128](bf16) @ [128 x 128](packed bf16) ----------------
// All row buffers padded to a multiple of 128 rows: NO per-lane guards anywhere.
// mode 0: out = A@W
// mode 1: out = silu(A@W + bias)
// mode 2: out = alpha * silu(A@W + bias) + resid
__global__ __launch_bounds__(256) void gemm128_wmma_kernel(
    const unsigned* __restrict__ Abf,      // [Npad x 64] dwords (2 bf16 each)
    const unsigned* __restrict__ Wpk,      // 8192 dwords, fragment layout
    const float* __restrict__ bias,
    const float* __restrict__ resid,
    const float* __restrict__ alpha_p, int alpha_idx,
    float* __restrict__ out, int mode) {
  int wave = threadIdx.x >> 5, lane = threadIdx.x & 31;
  int rowTile = (blockIdx.x * 8 + wave) * 16;

  int mlane = lane & 15;
  int hi = lane >> 4;               // lane half selects K-interleave / M-offset
  // A-matrix: M = lane&15 for both lane halves; row data 64 dwords = 16 uint4
  const uint4* arow = (const uint4*)(Abf + (size_t)(rowTile + mlane) * 64);

  v8f zero = {0.f, 0.f, 0.f, 0.f, 0.f, 0.f, 0.f, 0.f};
  v8f acc[8];
#pragma unroll
  for (int nt = 0; nt < 8; ++nt) acc[nt] = zero;

#pragma unroll
  for (int ks = 0; ks < 4; ++ks) {
    // A fragment (16x32 bf16): dwords {kb/2 .. kb/2+3} and {kb/2+8 .. kb/2+11},
    // kb = ks*32 + hi*8  ->  uint4 indices ks*4+hi and ks*4+hi+2
    BF8 afrag;
    uint4 a0 = arow[ks * 4 + hi];
    uint4 a1 = arow[ks * 4 + hi + 2];
    afrag.u[0] = a0.x; afrag.u[1] = a0.y; afrag.u[2] = a0.z; afrag.u[3] = a0.w;
    afrag.u[4] = a1.x; afrag.u[5] = a1.y; afrag.u[6] = a1.z; afrag.u[7] = a1.w;

    // preload all 8 B fragments for this k-step, then issue the 8 WMMAs
    BF8 bfrag[8];
#pragma unroll
    for (int nt = 0; nt < 8; ++nt) {
      const uint4* wp = (const uint4*)(Wpk + (((ks * 8 + nt) * 32 + lane) * 8));
      uint4 b0 = wp[0];
      uint4 b1 = wp[1];
      bfrag[nt].u[0] = b0.x; bfrag[nt].u[1] = b0.y; bfrag[nt].u[2] = b0.z; bfrag[nt].u[3] = b0.w;
      bfrag[nt].u[4] = b1.x; bfrag[nt].u[5] = b1.y; bfrag[nt].u[6] = b1.z; bfrag[nt].u[7] = b1.w;
    }
#pragma unroll
    for (int nt = 0; nt < 8; ++nt) {
      acc[nt] = __builtin_amdgcn_wmma_f32_16x16x32_bf16(
          false, afrag.v, false, bfrag[nt].v, (short)0, acc[nt], false, false);
    }
  }

  float alpha = (mode == 2) ? alpha_p[alpha_idx] : 0.0f;
  // C layout: VGPR r -> M = r + 8*hi, N = nt*16 + (lane&15)
#pragma unroll
  for (int nt = 0; nt < 8; ++nt) {
    int col = nt * 16 + mlane;
#pragma unroll
    for (int r = 0; r < 8; ++r) {
      size_t oi = (size_t)(rowTile + hi * 8 + r) * D_ + col;
      float y = acc[nt][r];
      if (mode == 0) {
        out[oi] = y;
      } else {
        y += bias[col];
        float s = silu_f(y);
        out[oi] = (mode == 2) ? alpha * s + resid[oi] : s;
      }
    }
  }
}

// ---------------- edge scatter: one wave per edge ----------------
__global__ __launch_bounds__(256) void scatter_kernel(
    const int* __restrict__ row_, const int* __restrict__ col_,
    const float* __restrict__ dinv, const float* __restrict__ t,
    float* __restrict__ agg, int nedges) {
  int wave = threadIdx.x >> 5, lane = threadIdx.x & 31;
  int e = blockIdx.x * 8 + wave;
  if (e >= nedges) return;
  int u = row_[e], v = col_[e];
  float w = dinv[u] * dinv[v];
  const float4* ts = (const float4*)(t + (size_t)u * D_);
  float4 m = ts[lane];
  float* av = agg + (size_t)v * D_ + lane * 4;
  atomicAdd(av + 0, w * m.x);
  atomicAdd(av + 1, w * m.y);
  atomicAdd(av + 2, w * m.z);
  atomicAdd(av + 3, w * m.w);
}

// ---------------- post conv: h = alpha*silu(agg + dinv^2*t + bias) + h ----------------
__global__ void post_conv_kernel(
    const float* __restrict__ agg, const float* __restrict__ t,
    const float* __restrict__ dinv, const float* __restrict__ bias,
    const float* __restrict__ alpha_p, int alpha_idx,
    float* __restrict__ h, int total) {
  int i = blockIdx.x * 256 + threadIdx.x;
  if (i >= total) return;
  int row = i >> 7, d = i & 127;
  float di = dinv[row];
  float y = agg[i] + di * di * t[i] + bias[d];
  h[i] = alpha_p[alpha_idx] * silu_f(y) + h[i];
}

// ---------------- final: logits = h @ Wf + bf, log_softmax (one wave per row) ----------------
__global__ __launch_bounds__(256) void final_kernel(
    const float* __restrict__ h, const float* __restrict__ Wf, const float* __restrict__ bf,
    float* __restrict__ out, int nrows, int nclass) {
  __shared__ float sh[8][D_];
  int wave = threadIdx.x >> 5, lane = threadIdx.x & 31;
  int row = blockIdx.x * 8 + wave;
  if (row >= nrows) return;
  const float* hr = h + (size_t)row * D_;
#pragma unroll
  for (int j = 0; j < 4; ++j) sh[wave][lane + 32 * j] = hr[lane + 32 * j];

  int c0 = lane;
  int c1 = lane + 32;
  bool has1 = c1 < nclass;
  float a0 = bf[c0];
  float a1 = has1 ? bf[c1] : 0.0f;
  int c1s = has1 ? c1 : 0;
  for (int d = 0; d < D_; ++d) {
    float hv = sh[wave][d];
    a0 += hv * Wf[d * nclass + c0];
    a1 += hv * Wf[d * nclass + c1s];
  }
  const float NEGBIG = -3.402823466e38f;
  float mx = fmaxf(a0, has1 ? a1 : NEGBIG);
#pragma unroll
  for (int m = 16; m; m >>= 1) mx = fmaxf(mx, __shfl_xor(mx, m, 32));
  float se = __expf(a0 - mx) + (has1 ? __expf(a1 - mx) : 0.0f);
#pragma unroll
  for (int m = 16; m; m >>= 1) se += __shfl_xor(se, m, 32);
  float lse = mx + logf(se);
  float* orow = out + (size_t)row * nclass;
  orow[c0] = a0 - lse;
  if (has1) orow[c1] = a1 - lse;
}

extern "C" void kernel_launch(void* const* d_in, const int* in_sizes, int n_in,
                              void* d_out, int out_size, void* d_ws, size_t ws_size,
                              hipStream_t stream) {
  const float* x      = (const float*)d_in[0];
  const int*   ei     = (const int*)d_in[1];
  const float* startW = (const float*)d_in[2];
  const float* startB = (const float*)d_in[3];
  const float* ln1g   = (const float*)d_in[4];
  const float* ln1b   = (const float*)d_in[5];
  const float* convW  = (const float*)d_in[6];
  const float* convB  = (const float*)d_in[7];
  const float* alphaG = (const float*)d_in[8];
  const float* ln2g   = (const float*)d_in[9];
  const float* ln2b   = (const float*)d_in[10];
  const float* ffwW   = (const float*)d_in[11];
  const float* ffwB   = (const float*)d_in[12];
  const float* alphaF = (const float*)d_in[13];
  const float* finW   = (const float*)d_in[14];
  const float* finB   = (const float*)d_in[15];

  const int N = in_sizes[0] / D_;
  const int E = in_sizes[1] / 2;
  const int C = in_sizes[15];
  const int L = 3;
  const int ND = N * D_;
  const int Npad = (N + 127) & ~127;        // pad rows so gemm has zero guards
  const int NDp = Npad * D_;

  // workspace layout (all row buffers Npad rows)
  char* ws = (char*)d_ws;
  size_t ndp4 = (size_t)NDp * 4;
  float*    h    = (float*)(ws);
  float*    t    = (float*)(ws + ndp4);
  float*    agg  = (float*)(ws + 2 * ndp4);
  unsigned* abf  = (unsigned*)(ws + 3 * ndp4);                   // NDp bf16 = NDp*2 bytes
  float*    deg  = (float*)(ws + 3 * ndp4 + ndp4 / 2);
  float*    dinv = (float*)(ws + 3 * ndp4 + ndp4 / 2 + (size_t)N * 4);
  unsigned* wpk  = (unsigned*)(ws + 3 * ndp4 + ndp4 / 2 + (size_t)N * 8); // 7 * 8192 dwords

  const int* erow = ei;       // edge_index[0]
  const int* ecol = ei + E;   // edge_index[1]

  dim3 blk(256);
  int gN    = (N + 255) / 256;
  int gE    = (E + 255) / 256;
  int gND   = (ND + 255) / 256;
  int gGemm = Npad / 128;            // 8 waves x 16 rows per block, no remainder
  int gRow8 = (N + 7) / 8;           // wave-per-row kernels
  int gEdge = (E + 7) / 8;           // wave-per-edge scatter

  // --- pack weights (bf16, WMMA fragment layout) ---
  pack_w_kernel<<<32, blk, 0, stream>>>(startW, wpk);
  for (int i = 0; i < L; ++i)
    pack_w_kernel<<<32, blk, 0, stream>>>(convW + (size_t)i * D_ * D_, wpk + (size_t)(1 + i) * 8192);
  for (int i = 0; i < L; ++i)
    pack_w_kernel<<<32, blk, 0, stream>>>(ffwW + (size_t)i * D_ * D_, wpk + (size_t)(4 + i) * 8192);

  // --- zero the bf16 activation pad rows once (they stay zero: LN writes only rows < N) ---
  int padDwords = (NDp - ND) / 2;    // bf16 pad region as dwords
  if (padDwords > 0)
    zero_u32_kernel<<<(padDwords + 255) / 256, blk, 0, stream>>>(abf + (size_t)ND / 2, padDwords);

  // --- gcn_norm ---
  deg_init_kernel<<<gN, blk, 0, stream>>>(deg, N);
  deg_count_kernel<<<gE, blk, 0, stream>>>(ecol, deg, E);
  dinv_kernel<<<gN, blk, 0, stream>>>(deg, dinv, N);

  // --- h = silu(x @ startW + startB) ---
  cvt_bf16_kernel<<<gND, blk, 0, stream>>>(x, (unsigned short*)abf, ND);
  gemm128_wmma_kernel<<<gGemm, blk, 0, stream>>>(abf, wpk, startB, nullptr, alphaG, 0, h, 1);

  for (int i = 0; i < L; ++i) {
    // GCN block
    ln_bf16_kernel<<<gRow8, blk, 0, stream>>>(h, ln1g + i * D_, ln1b + i * D_, (unsigned short*)abf, N);
    gemm128_wmma_kernel<<<gGemm, blk, 0, stream>>>(abf, wpk + (size_t)(1 + i) * 8192,
                                                   nullptr, nullptr, alphaG, i, t, 0);
    zero_f32_kernel<<<gND, blk, 0, stream>>>(agg, ND);
    scatter_kernel<<<gEdge, blk, 0, stream>>>(erow, ecol, dinv, t, agg, E);
    post_conv_kernel<<<gND, blk, 0, stream>>>(agg, t, dinv, convB + i * D_, alphaG, i, h, ND);
    // FFN block
    ln_bf16_kernel<<<gRow8, blk, 0, stream>>>(h, ln2g + i * D_, ln2b + i * D_, (unsigned short*)abf, N);
    gemm128_wmma_kernel<<<gGemm, blk, 0, stream>>>(abf, wpk + (size_t)(4 + i) * 8192,
                                                   ffwB + i * D_, h, alphaF, i, h, 2);
  }

  // --- final linear + log_softmax ---
  final_kernel<<<gRow8, blk, 0, stream>>>(h, finW, finB, (float*)d_out, N, C);
}
